// CARDModule_52390011077384
// MI455X (gfx1250) — matrix operand order repaired and checked by reading, not compiled, not run on hardware
//
#include <hip/hip_runtime.h>
#include <hip/hip_bf16.h>
#include <math.h>

#define NODES    32704
#define TREES    64
#define PERTREE  511

typedef __attribute__((ext_vector_type(16))) __bf16 v16bf;
typedef __attribute__((ext_vector_type(8)))  __bf16 v8bf;
typedef __attribute__((ext_vector_type(8)))  float  v8f;

__device__ __forceinline__ float bf2f(__bf16 b) {
    unsigned short s = __builtin_bit_cast(unsigned short, b);
    unsigned u = ((unsigned)s) << 16;
    return __builtin_bit_cast(float, u);
}
__device__ __forceinline__ __bf16 f2bf(float f) {
    unsigned u = __builtin_bit_cast(unsigned, f);
    u += 0x7FFFu + ((u >> 16) & 1u);           // round-to-nearest-even
    unsigned short s = (unsigned short)(u >> 16);
    return __builtin_bit_cast(__bf16, s);
}

// ---------------------------------------------------------------- convert
__global__ void cvt_f32_bf16(const float* __restrict__ s, __bf16* __restrict__ d, int n) {
    int i = blockIdx.x * blockDim.x + threadIdx.x;
    int stride = gridDim.x * blockDim.x;
    for (; i < n; i += stride) d[i] = f2bf(s[i]);
}

// ---------------------------------------------------------------- WMMA GEMM
// O[M x Ncol] = epi(A[M x K](bf16) * B[K x Ncol](bf16, row-major) + bias)
// epi==0: ReLU.  epi==1: identity for col<640, sigmoid for col>=640 (xou split).
// Block: 256 threads = 8 waves; tile 128(M) x 64(N); K step 32.
__launch_bounds__(256)
__global__ void gemm_bf16(const __bf16* __restrict__ A, int lda,
                          const __bf16* __restrict__ B,
                          const float* __restrict__ bias,
                          __bf16* __restrict__ O, int ldo, int colOff,
                          int M, int Ncol, int K, int epi)
{
    __shared__ __bf16 lA[128][40];   // rows padded: 80B stride -> conflict-free b128
    __shared__ __bf16 lB[64][40];    // n-major (transposed tile)

    const int tid   = threadIdx.x;
    const int mBase = blockIdx.x * 128;
    const int nBase = blockIdx.y * 64;
    const int wave  = tid >> 5;
    const int lane  = tid & 31;
    const int half  = lane >> 4;
    const int lr    = lane & 15;
    const int waveM = wave & 3;      // 4 waves along M (4*32 = 128)
    const int waveN = wave >> 2;     // 2 waves along N (2*32 = 64)

    v8f acc[2][2] = {};

    for (int k0 = 0; k0 < K; k0 += 32) {
        // ---- stage A: 128x32 (512 chunks of 8 bf16) ----
        #pragma unroll
        for (int c = 0; c < 2; ++c) {
            int ch  = tid + c * 256;
            int row = ch >> 2;
            int kc  = (ch & 3) * 8;
            int ar  = mBase + row;
            uint4 val = {0u, 0u, 0u, 0u};
            if (ar < M && (k0 + kc) < K)
                val = *(const uint4*)(A + (size_t)ar * lda + k0 + kc);
            *(uint4*)&lA[row][kc] = val;
        }
        // ---- stage B transposed: 32(K) x 64(N) -> lB[n][k] ----
        {
            int kk = tid >> 3;             // 0..31
            int nc = (tid & 7) * 8;        // 0..56
            if (k0 + kk < K) {
                v8bf bv = *(const v8bf*)(B + (size_t)(k0 + kk) * Ncol + nBase + nc);
                #pragma unroll
                for (int i = 0; i < 8; ++i) lB[nc + i][kk] = bv[i];
            } else {
                __bf16 z = __builtin_bit_cast(__bf16, (unsigned short)0);
                #pragma unroll
                for (int i = 0; i < 8; ++i) lB[nc + i][kk] = z;
            }
        }
        __syncthreads();

        // ---- fragments per documented VGPR layouts ----
        v16bf afrag[2], bfrag[2];
        #pragma unroll
        for (int mt = 0; mt < 2; ++mt) {
            int row = waveM * 32 + mt * 16 + lr;
            v8bf lo = *(const v8bf*)&lA[row][half * 8];        // K = half*8 + 0..7
            v8bf hi = *(const v8bf*)&lA[row][16 + half * 8];   // K = 16 + half*8 + 0..7
            #pragma unroll
            for (int i = 0; i < 8; ++i) { afrag[mt][i] = lo[i]; afrag[mt][8 + i] = hi[i]; }
        }
        #pragma unroll
        for (int nt = 0; nt < 2; ++nt) {
            int n = waveN * 32 + nt * 16 + lr;
            v8bf lo = *(const v8bf*)&lB[n][half * 8];          // K = half*8 + 0..7
            v8bf hi = *(const v8bf*)&lB[n][16 + half * 8];     // K = 16 + half*8 + 0..7
            #pragma unroll
            for (int i = 0; i < 8; ++i) { bfrag[nt][i] = lo[i]; bfrag[nt][8 + i] = hi[i]; }
        }
        #pragma unroll
        for (int mt = 0; mt < 2; ++mt)
            #pragma unroll
            for (int nt = 0; nt < 2; ++nt)
                acc[mt][nt] = __builtin_amdgcn_wmma_f32_16x16x32_bf16(
                    false, afrag[mt], false, bfrag[nt],
                    (short)0, acc[mt][nt], false, false);
        __syncthreads();
    }

    // ---- epilogue: bias + activation, store bf16 ----
    #pragma unroll
    for (int mt = 0; mt < 2; ++mt)
        #pragma unroll
        for (int nt = 0; nt < 2; ++nt) {
            int gcol = nBase + waveN * 32 + nt * 16 + lr;
            float bv = bias[gcol];
            #pragma unroll
            for (int v = 0; v < 8; ++v) {
                int grow = mBase + waveM * 32 + mt * 16 + half * 8 + v;
                if (grow < M) {
                    float val = acc[mt][nt][v] + bv;
                    if (epi == 0)            val = fmaxf(val, 0.f);
                    else if (gcol >= 640)    val = 1.f / (1.f + __expf(-val));
                    O[(size_t)grow * ldo + colOff + gcol] = f2bf(val);
                }
            }
        }
}

// ---------------------------------------------------------------- tree recurrence
// level n: nodes p in [2^(8-n)-1, 2^(9-n)-2] per tree; children 2p+1, 2p+2.
__global__ void tree_level(const __bf16* __restrict__ xou, const __bf16* __restrict__ xbf,
                           float* __restrict__ c, __bf16* __restrict__ hbf, int level)
{
    int cnt = 1 << (8 - level);
    int s   = cnt - 1;
    long long total = (long long)TREES * cnt * 640;
    long long idx = (long long)blockIdx.x * blockDim.x + threadIdx.x;
    if (idx >= total) return;
    int dim = (int)(idx % 640);
    long long t = idx / 640;
    int p    = s + (int)(t % cnt);
    int tree = (int)(t / cnt);
    int g    = tree * PERTREE + p;
    size_t base = (size_t)g * 1920;
    float xx = bf2f(xou[base + dim]);
    float f  = bf2f(xou[base + 640 + dim]);    // already sigmoid'ed
    float r  = bf2f(xou[base + 1280 + dim]);   // already sigmoid'ed
    float xv = bf2f(xbf[(size_t)g * 640 + dim]);
    float cs = 0.f;
    if (level > 0) {
        int c1 = tree * PERTREE + 2 * p + 1;
        cs = c[(size_t)c1 * 640 + dim] + c[(size_t)(c1 + 1) * 640 + dim];
    }
    float cv = f * cs + (1.f - f) * xx;
    c[(size_t)g * 640 + dim] = cv;
    float hv = r * tanhf(cv) + (1.f - r) * xv;
    hbf[(size_t)g * 640 + dim] = f2bf(hv);
}

// ---------------------------------------------------------------- output head
__global__ void out_head(const __bf16* __restrict__ g, const float* __restrict__ w_o2,
                         const float* __restrict__ b_o2, float* __restrict__ out, int n)
{
    int row  = (int)((blockIdx.x * (long long)blockDim.x + threadIdx.x) >> 5);
    int lane = threadIdx.x & 31;
    if (row >= n) return;
    float s = 0.f;
    for (int j = lane; j < 512; j += 32)
        s += bf2f(g[(size_t)row * 512 + j]) * w_o2[j];
    #pragma unroll
    for (int off = 16; off; off >>= 1) s += __shfl_xor(s, off, 32);
    if (lane == 0) out[row] = 1.f / (1.f + __expf(-(s + b_o2[0])));
}

// ---------------------------------------------------------------- launcher
extern "C" void kernel_launch(void* const* d_in, const int* in_sizes, int n_in,
                              void* d_out, int out_size, void* d_ws, size_t ws_size,
                              hipStream_t stream) {
    (void)in_sizes; (void)n_in; (void)out_size; (void)ws_size;
    const int N = NODES;

    const float* feat[5] = { (const float*)d_in[0], (const float*)d_in[1],
                             (const float*)d_in[2], (const float*)d_in[3],
                             (const float*)d_in[4] };
    const int fdim[5] = { 32, 64, 256, 128, 16 };
    const float* w1[5] = { (const float*)d_in[8],  (const float*)d_in[12],
                           (const float*)d_in[16], (const float*)d_in[20],
                           (const float*)d_in[24] };
    const float* b1[5] = { (const float*)d_in[9],  (const float*)d_in[13],
                           (const float*)d_in[17], (const float*)d_in[21],
                           (const float*)d_in[25] };
    const float* w2[5] = { (const float*)d_in[10], (const float*)d_in[14],
                           (const float*)d_in[18], (const float*)d_in[22],
                           (const float*)d_in[26] };
    const float* b2[5] = { (const float*)d_in[11], (const float*)d_in[15],
                           (const float*)d_in[19], (const float*)d_in[23],
                           (const float*)d_in[27] };
    const float* w_xou = (const float*)d_in[28];
    const float* b_xou = (const float*)d_in[29];
    const float* w_o1  = (const float*)d_in[30];
    const float* b_o1  = (const float*)d_in[31];
    const float* w_o2  = (const float*)d_in[32];
    const float* b_o2  = (const float*)d_in[33];

    float* out_p = (float*)d_out;          // N
    float* c_p   = (float*)d_out + N;      // N x 640

    // ---- workspace bump allocator (bf16 buffers, 256B aligned) ----
    char* w = (char*)d_ws;
    auto alloc = [&](size_t elems) -> __bf16* {
        __bf16* p = (__bf16*)w;
        w += ((elems * 2 + 255) / 256) * 256;
        return p;
    };
    __bf16* fb[5];  for (int i = 0; i < 5; ++i) fb[i]  = alloc((size_t)N * fdim[i]);
    __bf16* w1b[5]; for (int i = 0; i < 5; ++i) w1b[i] = alloc((size_t)fdim[i] * 128);
    __bf16* w2b[5]; for (int i = 0; i < 5; ++i) w2b[i] = alloc((size_t)128 * 128);
    __bf16* wxoub = alloc((size_t)640 * 1920);
    __bf16* wo1b  = alloc((size_t)640 * 512);
    __bf16* h1b   = alloc((size_t)N * 128);
    __bf16* xb    = alloc((size_t)N * 640);
    __bf16* xoub  = alloc((size_t)N * 1920);
    __bf16* hb    = alloc((size_t)N * 640);
    __bf16* gb    = alloc((size_t)N * 512);

    auto cvt = [&](const float* s, __bf16* d, size_t n) {
        int blocks = (int)((n + 255) / 256); if (blocks > 4096) blocks = 4096;
        cvt_f32_bf16<<<blocks, 256, 0, stream>>>(s, d, (int)n);
    };
    for (int i = 0; i < 5; ++i) {
        cvt(feat[i], fb[i],  (size_t)N * fdim[i]);
        cvt(w1[i],   w1b[i], (size_t)fdim[i] * 128);
        cvt(w2[i],   w2b[i], (size_t)128 * 128);
    }
    cvt(w_xou, wxoub, (size_t)640 * 1920);
    cvt(w_o1,  wo1b,  (size_t)640 * 512);

    auto gemm = [&](const __bf16* A, int lda, const __bf16* B, const float* bias,
                    __bf16* O, int ldo, int colOff, int Ncol, int K, int epi) {
        dim3 grid((N + 127) / 128, Ncol / 64);
        gemm_bf16<<<grid, 256, 0, stream>>>(A, lda, B, bias, O, ldo, colOff,
                                            N, Ncol, K, epi);
    };

    // five feature MLPs -> x (N x 640, column blocks of 128)
    for (int i = 0; i < 5; ++i) {
        gemm(fb[i], fdim[i], w1b[i], b1[i], h1b, 128, 0,        128, fdim[i], 0);
        gemm(h1b,   128,     w2b[i], b2[i], xb,  640, 128 * i,  128, 128,     0);
    }
    // xou = x @ w_xou + b ; sigmoid on cols >= 640
    gemm(xb, 640, wxoub, b_xou, xoub, 1920, 0, 1920, 640, 1);

    // tree recurrence, leaves (level 0) up to root (level 8)
    for (int lvl = 0; lvl <= 8; ++lvl) {
        long long total = (long long)TREES * (1 << (8 - lvl)) * 640;
        int blocks = (int)((total + 255) / 256);
        tree_level<<<blocks, 256, 0, stream>>>(xoub, xb, c_p, hb, lvl);
    }

    // output head
    gemm(hb, 640, wo1b, b_o1, gb, 512, 0, 512, 640, 0);
    out_head<<<(N * 32 + 255) / 256, 256, 0, stream>>>(gb, w_o2, b_o2, out_p, N);
}
